// Attention_10376640987812
// MI455X (gfx1250) — matrix operand order repaired and checked by reading, not compiled, not run on hardware
//
#include <hip/hip_runtime.h>

// Problem constants (fixed by the reference)
#define Bq   4
#define Nn   2048
#define Cc   768
#define Hh   12
#define HD   64
#define C3   2304
#define ROWS 8192          // B*N
#define QSCALE 0.125f      // 64^-0.5

typedef __bf16 v16bf __attribute__((ext_vector_type(16)));
typedef __bf16 v8bf  __attribute__((ext_vector_type(8)));
typedef float  v8f   __attribute__((ext_vector_type(8)));

// ---------------------------------------------------------------------------
// Fragment loaders (wave32 WMMA register layouts, CDNA5 ISA §7.12.2)
// A 16x32 bf16: lane L holds row M=L%16; lanes 0-15 carry K=[0..7]∪[16..23],
// lanes 16-31 carry K=[8..15]∪[24..31]  -> two contiguous 16B segments.
static __device__ inline v16bf load_a16(const __bf16* rowptr, unsigned lane) {
    unsigned hh = (lane >> 4) * 8;
    v8bf s0 = *(const v8bf*)(rowptr + hh);
    v8bf s1 = *(const v8bf*)(rowptr + 16 + hh);
    return __builtin_shufflevector(s0, s1, 0,1,2,3,4,5,6,7,8,9,10,11,12,13,14,15);
}
// B 32x16 bf16: lane L holds column N=L%16; lanes 0-15 carry K=[0..7]∪[16..23],
// lanes 16-31 carry K=[8..15]∪[24..31] -> two contiguous 16B segments per lane
// (requires K-major ("transposed") storage for the B operand).
static __device__ inline v16bf load_b16(const __bf16* colptr, unsigned lane) {
    unsigned hh = (lane >> 4) * 8;
    v8bf s0 = *(const v8bf*)(colptr + hh);
    v8bf s1 = *(const v8bf*)(colptr + 16 + hh);
    return __builtin_shufflevector(s0, s1, 0,1,2,3,4,5,6,7,8,9,10,11,12,13,14,15);
}
static __device__ inline v8f wmma_bf16(v16bf a, v16bf b, v8f c) {
    return __builtin_amdgcn_wmma_f32_16x16x32_bf16(false, a, false, b, (short)0, c, false, false);
}

// ---------------------------------------------------------------------------
// Conversion kernels
__global__ void cvt_x_kernel(const float* __restrict__ x, __bf16* __restrict__ xb, int n) {
    int i = blockIdx.x * blockDim.x + threadIdx.x;
    if (i < n) xb[i] = (__bf16)x[i];
}
// w [C, 3C] row-major -> wT [3C, C]
__global__ void cvt_wqkvT_kernel(const float* __restrict__ w, __bf16* __restrict__ wt) {
    int i = blockIdx.x * blockDim.x + threadIdx.x;          // i = j*C + k
    if (i < C3 * Cc) {
        int j = i / Cc, k = i - j * Cc;
        wt[i] = (__bf16)w[(size_t)k * C3 + j];
    }
}
// w [C, C] -> wT [C, C]
__global__ void cvt_wprojT_kernel(const float* __restrict__ w, __bf16* __restrict__ wt) {
    int i = blockIdx.x * blockDim.x + threadIdx.x;
    if (i < Cc * Cc) {
        int j = i / Cc, k = i - j * Cc;
        wt[i] = (__bf16)w[(size_t)k * Cc + j];
    }
}

// ---------------------------------------------------------------------------
// Double-buffered 32x64-per-wave GEMM core: acc[2][4] += A[32xK] * B[Kx64]
// ~176 live VGPRs -> kernels using this must allow >=2-waves/EU allocation
// (__launch_bounds__(256, 2)) so nothing spills to scratch.
static __device__ inline void gemm_core(const __bf16* __restrict__ arow0,
                                        const __bf16* __restrict__ arow1,
                                        const __bf16* __restrict__ Wt,
                                        size_t col0, unsigned lane,
                                        v8f acc[2][4]) {
    v16bf a0 = load_a16(arow0, lane);
    v16bf a1 = load_a16(arow1, lane);
    v16bf bf[4];
    #pragma unroll
    for (int t = 0; t < 4; ++t)
        bf[t] = load_b16(Wt + (col0 + (size_t)t * 16) * Cc, lane);
    for (int kb = 32; kb < Cc; kb += 32) {
        // prefetch k-step kb while consuming k-step kb-32
        v16bf na0 = load_a16(arow0 + kb, lane);
        v16bf na1 = load_a16(arow1 + kb, lane);
        v16bf nbf[4];
        #pragma unroll
        for (int t = 0; t < 4; ++t)
            nbf[t] = load_b16(Wt + (col0 + (size_t)t * 16) * Cc + kb, lane);
        #pragma unroll
        for (int t = 0; t < 4; ++t) {
            acc[0][t] = wmma_bf16(a0, bf[t], acc[0][t]);
            acc[1][t] = wmma_bf16(a1, bf[t], acc[1][t]);
        }
        a0 = na0; a1 = na1;
        #pragma unroll
        for (int t = 0; t < 4; ++t) bf[t] = nbf[t];
    }
    #pragma unroll
    for (int t = 0; t < 4; ++t) {   // peeled tail
        acc[0][t] = wmma_bf16(a0, bf[t], acc[0][t]);
        acc[1][t] = wmma_bf16(a1, bf[t], acc[1][t]);
    }
}

// QKV GEMM: Y[8192,2304] = Xb @ WqkvT^T + bias, scattered to Q/K/Vt (bf16).
__global__ void __launch_bounds__(256, 2)
qkv_gemm_kernel(const __bf16* __restrict__ X,
                const __bf16* __restrict__ Wt,   // [2304,768] K-major
                const float*  __restrict__ bias,
                __bf16* __restrict__ Qb,
                __bf16* __restrict__ Kb,
                __bf16* __restrict__ Vt) {
    unsigned lane = threadIdx.x & 31;
    unsigned w    = threadIdx.x >> 5;
    unsigned m0   = blockIdx.y * 256 + w * 32;
    unsigned n0   = blockIdx.x * 64;

    v8f acc[2][4] = {};
    const __bf16* arow0 = X + (size_t)(m0 + (lane & 15)) * Cc;
    gemm_core(arow0, arow0 + (size_t)16 * Cc, Wt, n0 + (lane & 15), lane, acc);

    #pragma unroll
    for (int t = 0; t < 4; ++t) {
        unsigned j = n0 + t * 16 + (lane & 15);       // column in [0, 2304)
        unsigned which = j / Cc;                      // 0=Q 1=K 2=V
        unsigned rem = j - which * Cc;
        unsigned head = rem >> 6, d = rem & 63;
        float bj = bias[j];
        #pragma unroll
        for (int i = 0; i < 2; ++i) {
            #pragma unroll
            for (int r = 0; r < 8; ++r) {
                unsigned row = m0 + 16 * i + r + 8 * (lane >> 4);
                unsigned bb = row >> 11, n = row & (Nn - 1);
                float val = acc[i][t][r] + bj;
                size_t bh = (size_t)bb * Hh + head;
                if (which == 0)      Qb[(bh * Nn + n) * HD + d] = (__bf16)(val * QSCALE);
                else if (which == 1) Kb[(bh * Nn + n) * HD + d] = (__bf16)val;
                else                 Vt[(bh * HD + d) * Nn + n] = (__bf16)val;
            }
        }
    }
}

// ---------------------------------------------------------------------------
// Flash attention: one wave per 16-query tile; keys in chunks of 32.
// V fragments loaded at chunk start, next-chunk K fragments prefetched right
// after the QK^T WMMAs, so the softmax VALU block covers their latency.
static __device__ inline void load_kfrags(const __bf16* __restrict__ Kh,
                                          int kc, unsigned lane, v16bf kf[4]) {
    const __bf16* kp0 = Kh + (size_t)(kc + (lane & 15)) * HD;
    const __bf16* kp1 = Kh + (size_t)(kc + 16 + (lane & 15)) * HD;
    kf[0] = load_b16(kp0, lane);
    kf[1] = load_b16(kp0 + 32, lane);
    kf[2] = load_b16(kp1, lane);
    kf[3] = load_b16(kp1 + 32, lane);
}

__global__ void __launch_bounds__(256, 2)
attn_kernel(const __bf16* __restrict__ Qb,
            const __bf16* __restrict__ Kb,
            const __bf16* __restrict__ Vt,
            __bf16* __restrict__ AO) {        // [B,N,C] bf16
    __shared__ __bf16 ldsP[8 * 16 * 32];
    unsigned lane = threadIdx.x & 31;
    unsigned w    = threadIdx.x >> 5;
    __bf16* lp    = ldsP + w * 512;

    unsigned bh    = blockIdx.y;                  // b*H + h
    unsigned bb    = bh / Hh, h = bh - bb * Hh;
    unsigned qbase = (blockIdx.x * 8 + w) * 16;

    const __bf16* Qh  = Qb + (size_t)bh * Nn * HD;
    const __bf16* Kh  = Kb + (size_t)bh * Nn * HD;
    const __bf16* Vth = Vt + (size_t)bh * HD * Nn;

    const __bf16* qrow = Qh + (size_t)(qbase + (lane & 15)) * HD;
    v16bf qa0 = load_a16(qrow, lane);
    v16bf qa1 = load_a16(qrow + 32, lane);

    v8f acc[4] = {};
    float mrow[8], lrow[8];
    #pragma unroll
    for (int r = 0; r < 8; ++r) { mrow[r] = -INFINITY; lrow[r] = 0.0f; }

    v16bf kf[4];
    load_kfrags(Kh, 0, lane, kf);

    for (int kc = 0; kc < Nn; kc += 32) {
        // V fragments for this chunk: needed only after softmax -> issue early
        v16bf vf[4];
        #pragma unroll
        for (int t = 0; t < 4; ++t)
            vf[t] = load_b16(Vth + (size_t)(t * 16 + (lane & 15)) * Nn + kc, lane);
        // S[16,32] = Q @ K^T  (4 WMMAs on prefetched K fragments)
        v8f s0 = {}, s1 = {};
        s0 = wmma_bf16(qa0, kf[0], s0);
        s0 = wmma_bf16(qa1, kf[1], s0);
        s1 = wmma_bf16(qa0, kf[2], s1);
        s1 = wmma_bf16(qa1, kf[3], s1);
        // prefetch next chunk's K fragments (clamped redundant load on last)
        int kn = (kc + 32 < Nn) ? kc + 32 : kc;
        load_kfrags(Kh, kn, lane, kf);
        // Online softmax (row reductions across 16-lane halves of the wave)
        #pragma unroll
        for (int r = 0; r < 8; ++r) {
            float cm = fmaxf(s0[r], s1[r]);
            #pragma unroll
            for (int msk = 8; msk >= 1; msk >>= 1) cm = fmaxf(cm, __shfl_xor(cm, msk, 16));
            float nm  = fmaxf(mrow[r], cm);
            float fac = __expf(mrow[r] - nm);
            float p0  = __expf(s0[r] - nm);
            float p1  = __expf(s1[r] - nm);
            float rs  = p0 + p1;
            #pragma unroll
            for (int msk = 8; msk >= 1; msk >>= 1) rs += __shfl_xor(rs, msk, 16);
            lrow[r] = lrow[r] * fac + rs;
            mrow[r] = nm;
            #pragma unroll
            for (int t = 0; t < 4; ++t) acc[t][r] *= fac;
            // stage P tile (C layout -> LDS, row-major 16x32)
            unsigned row = r + 8 * (lane >> 4);
            lp[row * 32 + (lane & 15)]      = (__bf16)p0;
            lp[row * 32 + 16 + (lane & 15)] = (__bf16)p1;
        }
        asm volatile("s_wait_dscnt 0" ::: "memory");   // wave-local LDS RAW fence
        v16bf pa = load_a16(lp + (lane & 15) * 32, lane);
        // O += P @ V  (4 WMMAs; V fragments have been in flight since chunk start)
        #pragma unroll
        for (int t = 0; t < 4; ++t)
            acc[t] = wmma_bf16(pa, vf[t], acc[t]);
    }
    // Normalize and write [B,N,C] bf16
    #pragma unroll
    for (int t = 0; t < 4; ++t) {
        #pragma unroll
        for (int r = 0; r < 8; ++r) {
            unsigned row = qbase + r + 8 * (lane >> 4);
            float val = acc[t][r] / lrow[r];
            AO[((size_t)bb * Nn + row) * Cc + h * HD + t * 16 + (lane & 15)] = (__bf16)val;
        }
    }
}

// ---------------------------------------------------------------------------
// Output projection: out[8192,768] f32 = AO @ WprojT^T + bias
__global__ void __launch_bounds__(256, 2)
proj_gemm_kernel(const __bf16* __restrict__ A,
                 const __bf16* __restrict__ Wt,   // [768,768] K-major
                 const float*  __restrict__ bias,
                 float* __restrict__ out) {
    unsigned lane = threadIdx.x & 31;
    unsigned w    = threadIdx.x >> 5;
    unsigned m0   = blockIdx.y * 256 + w * 32;
    unsigned n0   = blockIdx.x * 64;

    v8f acc[2][4] = {};
    const __bf16* arow0 = A + (size_t)(m0 + (lane & 15)) * Cc;
    gemm_core(arow0, arow0 + (size_t)16 * Cc, Wt, n0 + (lane & 15), lane, acc);

    #pragma unroll
    for (int t = 0; t < 4; ++t) {
        unsigned col = n0 + t * 16 + (lane & 15);
        float bj = bias[col];
        #pragma unroll
        for (int i = 0; i < 2; ++i) {
            #pragma unroll
            for (int r = 0; r < 8; ++r) {
                unsigned row = m0 + 16 * i + r + 8 * (lane >> 4);
                out[(size_t)row * Cc + col] = acc[i][t][r] + bj;
            }
        }
    }
}

// ---------------------------------------------------------------------------
extern "C" void kernel_launch(void* const* d_in, const int* in_sizes, int n_in,
                              void* d_out, int out_size, void* d_ws, size_t ws_size,
                              hipStream_t stream) {
    const float* x      = (const float*)d_in[0];
    const float* w_qkv  = (const float*)d_in[1];
    const float* b_qkv  = (const float*)d_in[2];
    const float* w_proj = (const float*)d_in[3];
    const float* b_proj = (const float*)d_in[4];
    float* out = (float*)d_out;

    char* ws = (char*)d_ws;
    size_t oXb = 0;
    size_t oWq = oXb + (size_t)ROWS * Cc * 2;     // Xb
    size_t oWp = oWq + (size_t)C3 * Cc * 2;       // WqkvT
    size_t oQ  = oWp + (size_t)Cc * Cc * 2;       // WprojT
    size_t oK  = oQ  + (size_t)ROWS * Cc * 2;     // Q
    size_t oVt = oK  + (size_t)ROWS * Cc * 2;     // K
    size_t oAO = oVt + (size_t)ROWS * Cc * 2;     // Vt
    __bf16* Xb  = (__bf16*)(ws + oXb);
    __bf16* WqT = (__bf16*)(ws + oWq);
    __bf16* WpT = (__bf16*)(ws + oWp);
    __bf16* Qb  = (__bf16*)(ws + oQ);
    __bf16* Kb  = (__bf16*)(ws + oK);
    __bf16* Vt  = (__bf16*)(ws + oVt);
    __bf16* AO  = (__bf16*)(ws + oAO);

    // 1) downconvert / transpose
    {
        int n = ROWS * Cc;
        cvt_x_kernel<<<(n + 255) / 256, 256, 0, stream>>>(x, Xb, n);
        int nq = C3 * Cc;
        cvt_wqkvT_kernel<<<(nq + 255) / 256, 256, 0, stream>>>(w_qkv, WqT);
        int np = Cc * Cc;
        cvt_wprojT_kernel<<<(np + 255) / 256, 256, 0, stream>>>(w_proj, WpT);
    }
    // 2) fused QKV projection (writes scaled Q, K, V^T)
    qkv_gemm_kernel<<<dim3(C3 / 64, ROWS / 256), 256, 0, stream>>>(Xb, WqT, b_qkv, Qb, Kb, Vt);
    // 3) flash attention
    attn_kernel<<<dim3(Nn / 128, Bq * Hh), 256, 0, stream>>>(Qb, Kb, Vt, AO);
    // 4) output projection + bias (f32 out)
    proj_gemm_kernel<<<dim3(Cc / 64, ROWS / 256), 256, 0, stream>>>(AO, WpT, b_proj, out);
}
